// Attention_35579509080321
// MI455X (gfx1250) — compile-verified, matrix operand order checked by reading, not run on hardware
//
#include <hip/hip_runtime.h>
#include <hip/hip_bf16.h>
#include <math.h>

// Problem constants (match reference)
#define B_    2
#define S_    2048
#define DIM_  2048
#define NH    16
#define NKV   4
#define HD    128
#define KVD   512          // NKV*HD
#define NTOK  (B_*S_)      // 4096

typedef __bf16 bf16;
typedef bf16  bf16x16 __attribute__((ext_vector_type(16)));
typedef float f32x8   __attribute__((ext_vector_type(8)));

// Load one 16-lane-striped WMMA operand fragment (A-layout or B-column-major
// layout): lane holds row (M or N) = lane%16; K elements split by lane half:
//   elems 0..7  -> K = half*8 + j
//   elems 8..15 -> K = 16 + half*8 + j
__device__ __forceinline__ bf16x16 ldfrag(const bf16* base, int hf) {
  union { bf16x16 v; uint4 q[2]; } u;
  u.q[0] = *(const uint4*)(base + hf * 8);
  u.q[1] = *(const uint4*)(base + 16 + hf * 8);
  return u.v;
}

// CDNA5 async global->LDS copy of 16 bytes (GLOBAL_LOAD_ASYNC_TO_LDS_B128).
// LDS destination address = low 32 bits of the generic pointer (the LDS
// aperture keeps the workgroup-relative offset in ADDR[31:0]).
__device__ __forceinline__ void async_copy16(void* lds_dst, const void* gsrc) {
  unsigned off = (unsigned)(size_t)lds_dst;
  asm volatile("global_load_async_to_lds_b128 %0, %1, off"
               :
               : "v"(off), "v"(gsrc)
               : "memory");
}

template <int N>
__device__ __forceinline__ void async_wait() {
#if __has_builtin(__builtin_amdgcn_s_wait_asynccnt)
  __builtin_amdgcn_s_wait_asynccnt(N);
#else
  asm volatile("s_wait_asynccnt %0" ::"i"(N) : "memory");
#endif
}

// ---------------------------------------------------------------------------
// f32 -> bf16 cast
// ---------------------------------------------------------------------------
__global__ void cast_bf16_kernel(const float* __restrict__ src,
                                 bf16* __restrict__ dst, int n) {
  int i = blockIdx.x * blockDim.x + threadIdx.x;
  if (i < n) dst[i] = (bf16)src[i];
}

// ---------------------------------------------------------------------------
// C[M,N] (f32) = A[M,K] (bf16, row major) @ W[N,K]^T (bf16, row major)
// Block: 256 threads (8 waves). Tile 256(M) x 128(N), K step 32, double-
// buffered LDS filled by async global->LDS loads. Wave w owns M rows
// [w*32, w*32+32). All 8 B fragments are preloaded into distinct registers
// so the 16 WMMAs per K step issue back-to-back behind one ds wait.
// ---------------------------------------------------------------------------
__global__ __launch_bounds__(256)
void gemm_bf16_kernel(const bf16* __restrict__ A, const bf16* __restrict__ W,
                      float* __restrict__ C, int M, int N, int K) {
  __shared__ bf16 As[2][256][40];  // 32 K + 8 pad (row = 80B, 16B aligned)
  __shared__ bf16 Bs[2][128][40];

  const int tid  = threadIdx.x;
  const int wave = tid >> 5;
  const int lane = tid & 31;
  const int hf   = lane >> 4;
  const int ln   = lane & 15;
  const int m0   = blockIdx.y * 256;
  const int n0   = blockIdx.x * 128;

  f32x8 acc0[8], acc1[8];
  f32x8 zero8 = {};
#pragma unroll
  for (int t = 0; t < 8; ++t) { acc0[t] = zero8; acc1[t] = zero8; }

  // Staging: A tile 256x32 -> thread owns row tid, 32 cols (4 x b128).
  //          B tile 128x32 -> thread owns row tid&127, 16 cols (2 x b128).
  const bf16* ag   = A + (size_t)(m0 + tid) * K;
  const int  brow  = tid & 127;
  const int  bhalf = tid >> 7;
  const bf16* wg   = W + (size_t)(n0 + brow) * K + bhalf * 16;

  auto prefetch = [&](int k0, int buf) {
    async_copy16(&As[buf][tid][0],  ag + k0);
    async_copy16(&As[buf][tid][8],  ag + k0 + 8);
    async_copy16(&As[buf][tid][16], ag + k0 + 16);
    async_copy16(&As[buf][tid][24], ag + k0 + 24);
    async_copy16(&Bs[buf][brow][bhalf * 16],     wg + k0);
    async_copy16(&Bs[buf][brow][bhalf * 16 + 8], wg + k0 + 8);
  };

  const int nk = K / 32;
  prefetch(0, 0);
  for (int kt = 0; kt < nk; ++kt) {
    const int c = kt & 1;
    if (kt + 1 < nk) {
      prefetch((kt + 1) * 32, c ^ 1);  // safe: buffer c^1 readers finished at
                                       // iteration kt-1's trailing barrier
      async_wait<6>();                 // 6 newer ops may stay in flight
    } else {
      async_wait<0>();
    }
    __syncthreads();  // whole tile visible to all waves

    bf16x16 a0 = ldfrag(&As[c][wave * 32 + ln][0], hf);
    bf16x16 a1 = ldfrag(&As[c][wave * 32 + 16 + ln][0], hf);
    bf16x16 bfr[8];
#pragma unroll
    for (int t = 0; t < 8; ++t) bfr[t] = ldfrag(&Bs[c][t * 16 + ln][0], hf);
#pragma unroll
    for (int t = 0; t < 8; ++t) {
      acc0[t] = __builtin_amdgcn_wmma_f32_16x16x32_bf16(
          false, a0, false, bfr[t], (short)0, acc0[t], false, false);
      acc1[t] = __builtin_amdgcn_wmma_f32_16x16x32_bf16(
          false, a1, false, bfr[t], (short)0, acc1[t], false, false);
    }
    __syncthreads();  // compute done before this buffer is refilled
  }

  // C/D layout: row = r + 8*half, col = lane%16 (per 16x16 tile)
#pragma unroll
  for (int t = 0; t < 8; ++t) {
#pragma unroll
    for (int r = 0; r < 8; ++r) {
      int n = n0 + t * 16 + ln;
      int ma = m0 + wave * 32 + r + 8 * hf;
      int mb = ma + 16;
      C[(size_t)ma * N + n] = acc0[t][r];
      C[(size_t)mb * N + n] = acc1[t][r];
    }
  }
}

// ---------------------------------------------------------------------------
// RoPE on Q and K (f32 in, bf16 out), plus V cast to bf16.
// ---------------------------------------------------------------------------
__global__ void rope_cast_kernel(const float* __restrict__ xq,
                                 const float* __restrict__ xk,
                                 const float* __restrict__ xv,
                                 const float* __restrict__ fcos,
                                 const float* __restrict__ fsin,
                                 bf16* __restrict__ qb, bf16* __restrict__ kb,
                                 bf16* __restrict__ vb) {
  int i = blockIdx.x * blockDim.x + threadIdx.x;
  const int total_q = NTOK * NH * (HD / 2);
  const int total_k = NTOK * NKV * (HD / 2);
  const int total_v = NTOK * KVD;

  if (i < total_q) {
    int p  = i & 63;
    int t1 = i >> 6;
    int h  = t1 & (NH - 1);
    int bs = t1 >> 4;  // b*S + s
    int s  = bs & (S_ - 1);
    float c  = fcos[s * 64 + p];
    float sn = fsin[s * 64 + p];
    size_t base = (size_t)bs * DIM_ + h * HD + 2 * p;
    float re = xq[base], im = xq[base + 1];
    qb[base]     = (bf16)(re * c - im * sn);
    qb[base + 1] = (bf16)(re * sn + im * c);
  }
  if (i < total_k) {
    int p  = i & 63;
    int t1 = i >> 6;
    int h  = t1 & (NKV - 1);
    int bs = t1 >> 2;
    int s  = bs & (S_ - 1);
    float c  = fcos[s * 64 + p];
    float sn = fsin[s * 64 + p];
    size_t base = (size_t)bs * KVD + h * HD + 2 * p;
    float re = xk[base], im = xk[base + 1];
    kb[base]     = (bf16)(re * c - im * sn);
    kb[base + 1] = (bf16)(re * sn + im * c);
  }
  if (i < total_v) vb[i] = (bf16)xv[i];
}

// ---------------------------------------------------------------------------
// Flash attention, causal, GQA (head h reads kv head h/4).
// Grid: (S/128, NH, B). Block: 256 threads = 8 waves; wave owns 16 q rows.
// K tile staged via async global->LDS; V staged transposed through registers.
// ---------------------------------------------------------------------------
__global__ __launch_bounds__(256)
void flash_attn_kernel(const bf16* __restrict__ qb, const bf16* __restrict__ kb,
                       const bf16* __restrict__ vb, bf16* __restrict__ ob) {
  __shared__ bf16 Kt[64][136];    // [kv][d], 128 + 8 pad
  __shared__ bf16 VtT[128][72];   // [d][kv], 64 + 8 pad
  __shared__ bf16 Pt[128][72];    // [m][kv] probabilities, per-wave rows

  const int tid  = threadIdx.x;
  const int wave = tid >> 5;
  const int lane = tid & 31;
  const int hf   = lane >> 4;
  const int ln   = lane & 15;
  const int q0   = blockIdx.x * 128;
  const int h    = blockIdx.y;
  const int b    = blockIdx.z;
  const int kvh  = h >> 2;  // N_REP = 4

  // Q fragments: 4 K-chunks of 32 over d=128, loaded straight from global in
  // A-fragment layout (lane holds row m = lane%16).
  const bf16* qrow =
      qb + (size_t)(b * S_ + q0 + wave * 16 + ln) * DIM_ + h * HD;
  bf16x16 qf[4];
#pragma unroll
  for (int c = 0; c < 4; ++c) qf[c] = ldfrag(qrow + c * 32, hf);

  f32x8 o[8];
  f32x8 zero8 = {};
#pragma unroll
  for (int t = 0; t < 8; ++t) o[t] = zero8;
  float mi[8], li[8];
#pragma unroll
  for (int r = 0; r < 8; ++r) { mi[r] = -3.0e38f; li[r] = 0.0f; }

  const float scale = 0.08838834764831845f;  // 1/sqrt(128)
  const int nkv = q0 / 64 + 2;               // causal: kv blocks <= q block end

  for (int j = 0; j < nkv; ++j) {
    const int kv0 = j * 64;
    // ---- stage K (async -> LDS) and transposed V (via registers) ----
    {
      int row = tid >> 2;  // 0..63 kv row
      int q4  = tid & 3;   // 32-col quarter
      const bf16* kg =
          kb + (size_t)(b * S_ + kv0 + row) * KVD + kvh * HD + q4 * 32;
      const bf16* vg =
          vb + (size_t)(b * S_ + kv0 + row) * KVD + kvh * HD + q4 * 32;
      union { uint4 q[4]; bf16 e[32]; } vv;
      vv.q[0] = *(const uint4*)(vg);
      vv.q[1] = *(const uint4*)(vg + 8);
      vv.q[2] = *(const uint4*)(vg + 16);
      vv.q[3] = *(const uint4*)(vg + 24);
      __syncthreads();  // all waves done reading previous tiles
      async_copy16(&Kt[row][q4 * 32],      kg);
      async_copy16(&Kt[row][q4 * 32 + 8],  kg + 8);
      async_copy16(&Kt[row][q4 * 32 + 16], kg + 16);
      async_copy16(&Kt[row][q4 * 32 + 24], kg + 24);
#pragma unroll
      for (int e = 0; e < 32; ++e) VtT[q4 * 32 + e][row] = vv.e[e];
      async_wait<0>();
      __syncthreads();  // staging visible
    }

    // ---- scores: S = Q @ K^T ----
    // Per d-chunk: preload 4 K fragments (distinct regs), then 4 independent
    // WMMAs into the 4 score accumulators.
    f32x8 sc[4];
#pragma unroll
    for (int t = 0; t < 4; ++t) sc[t] = zero8;
#pragma unroll
    for (int c = 0; c < 4; ++c) {
      bf16x16 kf[4];
#pragma unroll
      for (int t = 0; t < 4; ++t)
        kf[t] = ldfrag(&Kt[t * 16 + ln][c * 32], hf);
#pragma unroll
      for (int t = 0; t < 4; ++t)
        sc[t] = __builtin_amdgcn_wmma_f32_16x16x32_bf16(
            false, qf[c], false, kf[t], (short)0, sc[t], false, false);
    }

    // ---- online softmax (rows live across the 16 lanes of each half) ----
#pragma unroll
    for (int r = 0; r < 8; ++r) {
      const int qg = q0 + wave * 16 + r + 8 * hf;  // global q row
      float sv[4];
      float rmax = -3.0e38f;
#pragma unroll
      for (int t = 0; t < 4; ++t) {
        int kcol = kv0 + t * 16 + ln;
        float v = sc[t][r] * scale + (kcol > qg ? -1.0e9f : 0.0f);
        sv[t] = v;
        rmax = fmaxf(rmax, v);
      }
#pragma unroll
      for (int d = 1; d < 16; d <<= 1)
        rmax = fmaxf(rmax, __shfl_xor(rmax, d, 32));
      float mnew  = fmaxf(mi[r], rmax);
      float alpha = __expf(mi[r] - mnew);
      float rsum  = 0.0f;
#pragma unroll
      for (int t = 0; t < 4; ++t) {
        float p = __expf(sv[t] - mnew);
        rsum += p;
        Pt[wave * 16 + r + 8 * hf][t * 16 + ln] = (bf16)p;
      }
#pragma unroll
      for (int d = 1; d < 16; d <<= 1) rsum += __shfl_xor(rsum, d, 32);
      li[r] = li[r] * alpha + rsum;
      mi[r] = mnew;
#pragma unroll
      for (int t = 0; t < 8; ++t) o[t][r] = o[t][r] * alpha;
    }
    __syncthreads();  // Pt writes visible before fragment reloads

    // ---- O += P @ V : groups of 4 independent WMMAs per P fragment ----
    bf16x16 pf0 = ldfrag(&Pt[wave * 16 + ln][0], hf);
    bf16x16 pf1 = ldfrag(&Pt[wave * 16 + ln][32], hf);
#pragma unroll
    for (int tg = 0; tg < 2; ++tg) {
      bf16x16 vf[4];
#pragma unroll
      for (int t = 0; t < 4; ++t)
        vf[t] = ldfrag(&VtT[(tg * 4 + t) * 16 + ln][0], hf);
#pragma unroll
      for (int t = 0; t < 4; ++t)
        o[tg * 4 + t] = __builtin_amdgcn_wmma_f32_16x16x32_bf16(
            false, pf0, false, vf[t], (short)0, o[tg * 4 + t], false, false);
#pragma unroll
      for (int t = 0; t < 4; ++t)
        vf[t] = ldfrag(&VtT[(tg * 4 + t) * 16 + ln][32], hf);
#pragma unroll
      for (int t = 0; t < 4; ++t)
        o[tg * 4 + t] = __builtin_amdgcn_wmma_f32_16x16x32_bf16(
            false, pf1, false, vf[t], (short)0, o[tg * 4 + t], false, false);
    }
  }

  // ---- epilogue: divide by softmax sum, write bf16 (feeds output GEMM) ----
  bf16* orow = ob + (size_t)(b * S_ + q0 + wave * 16) * DIM_ + h * HD;
#pragma unroll
  for (int t = 0; t < 8; ++t) {
#pragma unroll
    for (int r = 0; r < 8; ++r) {
      int m = r + 8 * hf;
      float val = o[t][r] / li[r];
      orow[(size_t)m * DIM_ + t * 16 + ln] = (bf16)val;
    }
  }
}

// ---------------------------------------------------------------------------
// Host-side orchestration
// ---------------------------------------------------------------------------
extern "C" void kernel_launch(void* const* d_in, const int* in_sizes, int n_in,
                              void* d_out, int out_size, void* d_ws,
                              size_t ws_size, hipStream_t stream) {
  (void)in_sizes; (void)n_in; (void)out_size; (void)ws_size;

  const float* x   = (const float*)d_in[0];
  const float* wq  = (const float*)d_in[1];
  const float* wk  = (const float*)d_in[2];
  const float* wv  = (const float*)d_in[3];
  const float* wo  = (const float*)d_in[4];
  const float* fco = (const float*)d_in[5];
  const float* fsi = (const float*)d_in[6];
  // d_in[7] (mask) unused: causal mask computed inline, identical to reference.
  float* out = (float*)d_out;

  char* ws = (char*)d_ws;
  size_t off = 0;
  auto alloc = [&](size_t bytes) -> void* {
    void* p = ws + off;
    off = (off + bytes + 255) & ~(size_t)255;
    return p;
  };

  bf16*  xb  = (bf16*)alloc((size_t)NTOK * DIM_ * 2);
  bf16*  wqb = (bf16*)alloc((size_t)DIM_ * DIM_ * 2);
  bf16*  wkb = (bf16*)alloc((size_t)KVD * DIM_ * 2);
  bf16*  wvb = (bf16*)alloc((size_t)KVD * DIM_ * 2);
  bf16*  wob = (bf16*)alloc((size_t)DIM_ * DIM_ * 2);
  float* xqf = (float*)alloc((size_t)NTOK * DIM_ * 4);
  float* xkf = (float*)alloc((size_t)NTOK * KVD * 4);
  float* xvf = (float*)alloc((size_t)NTOK * KVD * 4);
  bf16*  qbb = (bf16*)alloc((size_t)NTOK * DIM_ * 2);
  bf16*  kbb = (bf16*)alloc((size_t)NTOK * KVD * 2);
  bf16*  vbb = (bf16*)alloc((size_t)NTOK * KVD * 2);
  bf16*  aob = (bf16*)alloc((size_t)NTOK * DIM_ * 2);

  const dim3 blk(256);

  // Casts to bf16
  cast_bf16_kernel<<<(NTOK * DIM_) / 256, blk, 0, stream>>>(x, xb, NTOK * DIM_);
  cast_bf16_kernel<<<(DIM_ * DIM_) / 256, blk, 0, stream>>>(wq, wqb, DIM_ * DIM_);
  cast_bf16_kernel<<<(KVD * DIM_) / 256, blk, 0, stream>>>(wk, wkb, KVD * DIM_);
  cast_bf16_kernel<<<(KVD * DIM_) / 256, blk, 0, stream>>>(wv, wvb, KVD * DIM_);
  cast_bf16_kernel<<<(DIM_ * DIM_) / 256, blk, 0, stream>>>(wo, wob, DIM_ * DIM_);

  // QKV projections (M = NTOK = 4096 is a multiple of the 256-row tile)
  gemm_bf16_kernel<<<dim3(DIM_ / 128, NTOK / 256), blk, 0, stream>>>(
      xb, wqb, xqf, NTOK, DIM_, DIM_);
  gemm_bf16_kernel<<<dim3(KVD / 128, NTOK / 256), blk, 0, stream>>>(
      xb, wkb, xkf, NTOK, KVD, DIM_);
  gemm_bf16_kernel<<<dim3(KVD / 128, NTOK / 256), blk, 0, stream>>>(
      xb, wvb, xvf, NTOK, KVD, DIM_);

  // RoPE + casts
  const int nq = NTOK * NH * (HD / 2);
  rope_cast_kernel<<<nq / 256, blk, 0, stream>>>(xqf, xkf, xvf, fco, fsi, qbb,
                                                 kbb, vbb);

  // Attention
  flash_attn_kernel<<<dim3(S_ / 128, NH, B_), blk, 0, stream>>>(qbb, kbb, vbb,
                                                                aob);

  // Output projection -> f32 result
  gemm_bf16_kernel<<<dim3(DIM_ / 128, NTOK / 256), blk, 0, stream>>>(
      aob, wob, out, NTOK, DIM_, DIM_);
}
